// SAGE_35588099015580
// MI455X (gfx1250) — matrix operand order repaired and checked by reading, not compiled, not run on hardware
//
#include <hip/hip_runtime.h>

#define NFEAT 128
#define BN_EPS 1e-5f

typedef __attribute__((ext_vector_type(16))) __bf16 v16bf;
typedef __attribute__((ext_vector_type(8)))  __bf16 v8bf;
typedef __attribute__((ext_vector_type(8)))  float  v8f;

__device__ __forceinline__ unsigned short f2bf(float f) {
    unsigned u = __float_as_uint(f);
    unsigned r = ((u >> 16) & 1u) + 0x7fffu;   // round-to-nearest-even
    return (unsigned short)((u + r) >> 16);
}
__device__ __forceinline__ float bf2f(unsigned short h) {
    return __uint_as_float(((unsigned)h) << 16);
}

// ---------------------------------------------------------------- utilities
__global__ void zero_f32_kernel(float* __restrict__ p, long long n) {
    long long i = blockIdx.x * (long long)blockDim.x + threadIdx.x;
    long long stride = (long long)gridDim.x * blockDim.x;
    for (; i < n; i += stride) p[i] = 0.0f;
}

__global__ void f2bf_kernel(const float* __restrict__ in,
                            unsigned short* __restrict__ out, long long n) {
    long long i = blockIdx.x * (long long)blockDim.x + threadIdx.x;
    long long stride = (long long)gridDim.x * blockDim.x;
    for (; i < n; i += stride) out[i] = f2bf(in[i]);
}

// agg[i] / max(deg,1) -> bf16
__global__ void norm_bf_kernel(const float* __restrict__ agg,
                               const float* __restrict__ deg,
                               unsigned short* __restrict__ out, long long n) {
    long long i = blockIdx.x * (long long)blockDim.x + threadIdx.x;
    long long stride = (long long)gridDim.x * blockDim.x;
    for (; i < n; i += stride) {
        int row = (int)(i >> 7);
        float r = 1.0f / fmaxf(deg[row], 1.0f);
        out[i] = f2bf(agg[i] * r);
    }
}

// ---------------------------------------------------------------- scatter
// thread -> (edge, 4-feature chunk); 32 chunks per edge
__global__ __launch_bounds__(256)
void scatter_f32_kernel(const long long* __restrict__ ei,
                        const float* __restrict__ H,
                        float* __restrict__ agg, float* __restrict__ deg, int nE) {
    long long gid = blockIdx.x * (long long)blockDim.x + threadIdx.x;
    long long e = gid >> 5;
    if (e >= nE) return;
    int c = (int)(gid & 31) << 2;
    int s = (int)ei[e];
    int d = (int)ei[(long long)nE + e];
    const float4 v = *(const float4*)(H + (size_t)s * NFEAT + c);
    float* a = agg + (size_t)d * NFEAT + c;
    unsafeAtomicAdd(a + 0, v.x);
    unsafeAtomicAdd(a + 1, v.y);
    unsafeAtomicAdd(a + 2, v.z);
    unsafeAtomicAdd(a + 3, v.w);
    if (c == 0) unsafeAtomicAdd(deg + d, 1.0f);
}

__global__ __launch_bounds__(256)
void scatter_bf16_kernel(const long long* __restrict__ ei,
                         const unsigned short* __restrict__ H,
                         float* __restrict__ agg, int nE) {
    long long gid = blockIdx.x * (long long)blockDim.x + threadIdx.x;
    long long e = gid >> 5;
    if (e >= nE) return;
    int c = (int)(gid & 31) << 2;
    int s = (int)ei[e];
    int d = (int)ei[(long long)nE + e];
    unsigned long long pk = *(const unsigned long long*)(H + (size_t)s * NFEAT + c);
    float* a = agg + (size_t)d * NFEAT + c;
    unsafeAtomicAdd(a + 0, bf2f((unsigned short)(pk)));
    unsafeAtomicAdd(a + 1, bf2f((unsigned short)(pk >> 16)));
    unsafeAtomicAdd(a + 2, bf2f((unsigned short)(pk >> 32)));
    unsafeAtomicAdd(a + 3, bf2f((unsigned short)(pk >> 48)));
}

// ---------------------------------------------------------------- WMMA GEMM
// out[row, col] = RELU?( A@WlB^T + (HAS_H ? H@WrB^T : 0) + bias )
// One wave per 16x16 tile; 8 waves/block cover 128 cols; A/H row-major bf16,
// W row-major [128 x 128] bf16 (so W^T columns are contiguous W rows).
// Compile-time config -> branch-free K loop of back-to-back v_wmma.
template <bool HAS_H, bool OUT_F, bool OUT_B, bool RELU>
__global__ __launch_bounds__(256)
void sage_wmma_gemm_kernel(const unsigned short* __restrict__ A,
                           const unsigned short* __restrict__ H,
                           const unsigned short* __restrict__ WlB,
                           const unsigned short* __restrict__ WrB,
                           const float* __restrict__ bias,
                           float* __restrict__ outF,
                           unsigned short* __restrict__ outB,
                           int n) {
    const int lane = threadIdx.x & 31;
    const int wave = threadIdx.x >> 5;
    const int row0 = blockIdx.x * 16;
    const int col0 = wave * 16;
    const int r    = lane & 15;
    const int hi   = lane >> 4;

    int arow = row0 + r;
    if (arow >= n) arow = n - 1;                 // clamp for ragged tail
    const size_t abase = (size_t)arow * NFEAT;
    const size_t wbase = (size_t)(col0 + r) * NFEAT;

    v8f c = {};
#pragma unroll
    for (int k0 = 0; k0 < NFEAT; k0 += 32) {
        // A fragment: lane holds row (l&15); 16 contiguous K at k0+hi*16
        union { v16bf v; v8bf h[2]; } ua;
        const unsigned short* pa = A + abase + k0 + hi * 16;
        ua.h[0] = *(const v8bf*)(pa);
        ua.h[1] = *(const v8bf*)(pa + 8);
        // B fragment: lane holds column col0+(l&15) = W row; 16 contiguous K at k0+hi*16
        v16bf b = *(const v16bf*)(WlB + wbase + k0 + hi * 16);
        c = __builtin_amdgcn_wmma_f32_16x16x32_bf16(false, ua.v, false, b,
                                                    (short)0, c, false, false);
        if (HAS_H) {
            union { v16bf v; v8bf h[2]; } uh;
            const unsigned short* ph = H + abase + k0 + hi * 16;
            uh.h[0] = *(const v8bf*)(ph);
            uh.h[1] = *(const v8bf*)(ph + 8);
            v16bf b2 = *(const v16bf*)(WrB + wbase + k0 + hi * 16);
            c = __builtin_amdgcn_wmma_f32_16x16x32_bf16(false, uh.v, false, b2,
                                                        (short)0, c, false, false);
        }
    }

    const float bv = bias[col0 + r];
#pragma unroll
    for (int v = 0; v < 8; ++v) {
        int row = row0 + v + hi * 8;             // C/D layout: VGPR v, lane-half hi
        if (row < n) {
            float f = c[v] + bv;
            if (RELU) f = fmaxf(f, 0.0f);
            size_t idx = (size_t)row * NFEAT + col0 + r;
            if (OUT_F) outF[idx] = f;
            if (OUT_B) outB[idx] = f2bf(f);
        }
    }
}

// ---------------------------------------------------------------- batchnorm
__global__ __launch_bounds__(128)
void bn_stats_kernel(const float* __restrict__ h, float* __restrict__ sums, int n) {
    int col = threadIdx.x;                       // 128 cols
    float s = 0.0f, s2 = 0.0f;
    for (int row = blockIdx.x; row < n; row += gridDim.x) {
        float v = h[(size_t)row * NFEAT + col];
        s += v; s2 += v * v;
    }
    unsafeAtomicAdd(&sums[col], s);
    unsafeAtomicAdd(&sums[NFEAT + col], s2);
}

__global__ __launch_bounds__(128)
void bn_apply_kernel(const float* __restrict__ h, const float* __restrict__ sums,
                     const float* __restrict__ gamma, const float* __restrict__ beta,
                     unsigned short* __restrict__ out, int n) {
    int col = threadIdx.x;
    float inv_n = 1.0f / (float)n;
    float mu  = sums[col] * inv_n;
    float var = sums[NFEAT + col] * inv_n - mu * mu;   // biased variance
    float a = gamma[col] * rsqrtf(var + BN_EPS);
    float b = beta[col] - mu * a;
    for (int row = blockIdx.x; row < n; row += gridDim.x) {
        size_t i = (size_t)row * NFEAT + col;
        out[i] = f2bf(h[i] * a + b);
    }
}

// ---------------------------------------------------------------- launcher
extern "C" void kernel_launch(void* const* d_in, const int* in_sizes, int n_in,
                              void* d_out, int out_size, void* d_ws, size_t ws_size,
                              hipStream_t stream) {
    const float*     x     = (const float*)d_in[0];
    const long long* ei    = (const long long*)d_in[1];   // int64 [2, E]
    const float*     Wl1   = (const float*)d_in[2];
    const float*     bl1   = (const float*)d_in[3];
    const float*     Wr1   = (const float*)d_in[4];
    const float*     Wl2   = (const float*)d_in[5];
    const float*     bl2   = (const float*)d_in[6];
    const float*     Wr2   = (const float*)d_in[7];
    const float*     gamma = (const float*)d_in[8];
    const float*     beta  = (const float*)d_in[9];
    const float*     fcW   = (const float*)d_in[10];
    const float*     fcb   = (const float*)d_in[11];

    const int N = in_sizes[0] / NFEAT;
    const int E = in_sizes[1] / 2;
    const long long NF = (long long)N * NFEAT;

    // workspace carving (256B aligned)
    char* ws = (char*)d_ws;
    size_t off = 0;
    auto alloc = [&](size_t bytes) -> void* {
        void* p = ws + off;
        off = (off + bytes + 255) & ~(size_t)255;
        return p;
    };
    float*          aggF  = (float*)alloc((size_t)NF * 4);   // also reused as h2 (f32)
    float*          deg   = (float*)alloc((size_t)N * 4);
    float*          sums  = (float*)alloc(2 * NFEAT * 4);
    unsigned short* xbf   = (unsigned short*)alloc((size_t)NF * 2); // reused as bn output
    unsigned short* aggbf = (unsigned short*)alloc((size_t)NF * 2);
    unsigned short* h1bf  = (unsigned short*)alloc((size_t)NF * 2);
    unsigned short* wbf   = (unsigned short*)alloc(5 * 16384 * 2);

    const int TB = 256;
    const long long sthreads = (long long)E * 32;
    const int sblocks = (int)((sthreads + TB - 1) / TB);
    const int gblocks = (N + 15) / 16;

    // init + precision conversion
    zero_f32_kernel<<<2048, TB, 0, stream>>>(aggF, NF);
    zero_f32_kernel<<<256,  TB, 0, stream>>>(deg, N);
    zero_f32_kernel<<<1,    TB, 0, stream>>>(sums, 2 * NFEAT);
    f2bf_kernel<<<4096, TB, 0, stream>>>(x,   xbf,             NF);
    f2bf_kernel<<<64,   TB, 0, stream>>>(Wl1, wbf + 0 * 16384, 16384);
    f2bf_kernel<<<64,   TB, 0, stream>>>(Wr1, wbf + 1 * 16384, 16384);
    f2bf_kernel<<<64,   TB, 0, stream>>>(Wl2, wbf + 2 * 16384, 16384);
    f2bf_kernel<<<64,   TB, 0, stream>>>(Wr2, wbf + 3 * 16384, 16384);
    f2bf_kernel<<<64,   TB, 0, stream>>>(fcW, wbf + 4 * 16384, 16384);

    // ---- SAGE layer 1: h1 = relu(mean_agg(x) @ Wl1^T + bl1 + x @ Wr1^T) -> bf16
    scatter_f32_kernel<<<sblocks, TB, 0, stream>>>(ei, x, aggF, deg, E);
    norm_bf_kernel<<<4096, TB, 0, stream>>>(aggF, deg, aggbf, NF);
    sage_wmma_gemm_kernel<true, false, true, true><<<gblocks, TB, 0, stream>>>(
        aggbf, xbf, wbf + 0 * 16384, wbf + 1 * 16384, bl1,
        nullptr, h1bf, N);

    // ---- SAGE layer 2: h2 = relu(mean_agg(h1) @ Wl2^T + bl2 + h1 @ Wr2^T) -> f32
    zero_f32_kernel<<<2048, TB, 0, stream>>>(aggF, NF);
    scatter_bf16_kernel<<<sblocks, TB, 0, stream>>>(ei, h1bf, aggF, E);
    norm_bf_kernel<<<4096, TB, 0, stream>>>(aggF, deg, aggbf, NF);
    sage_wmma_gemm_kernel<true, true, false, true><<<gblocks, TB, 0, stream>>>(
        aggbf, h1bf, wbf + 2 * 16384, wbf + 3 * 16384, bl2,
        aggF /* h2 f32 (reuse) */, nullptr, N);

    // ---- BatchNorm (training stats, biased var) -> bf16
    bn_stats_kernel<<<1024, 128, 0, stream>>>(aggF, sums, N);
    bn_apply_kernel<<<2048, 128, 0, stream>>>(aggF, sums, gamma, beta, xbf, N);

    // ---- final FC: out = h2n @ fcW^T + fcb (f32 output)
    sage_wmma_gemm_kernel<false, true, false, false><<<gblocks, TB, 0, stream>>>(
        xbf, nullptr, wbf + 4 * 16384, nullptr, fcb,
        (float*)d_out, nullptr, N);
}
